// MultiHeadSelfAttentionCausal_46617575031070
// MI455X (gfx1250) — matrix operand order repaired and checked by reading, not compiled, not run on hardware
//
#include <hip/hip_runtime.h>
#include <hip/hip_bf16.h>

typedef _Float16 f16;
typedef __attribute__((ext_vector_type(4)))  _Float16 v4h;
typedef __attribute__((ext_vector_type(8)))  _Float16 v8h;
typedef __attribute__((ext_vector_type(16))) _Float16 v16h;
typedef __attribute__((ext_vector_type(8)))  float    v8f;
typedef __attribute__((ext_vector_type(4)))  unsigned u32x4;
typedef __attribute__((ext_vector_type(4)))  int      i32x4;
typedef __attribute__((ext_vector_type(8)))  int      i32x8;

#define EMBED 1024
#define NHEAD 16
#define HDIM  64
#define SEQ   2048
#define BATCH 4
#define QKV3  3072
#define MTOT  (BATCH * SEQ)   // 8192 rows

#if __has_builtin(__builtin_amdgcn_tensor_load_to_lds) && \
    __has_builtin(__builtin_amdgcn_s_wait_tensorcnt)
#define HAVE_TDM 1
#else
#define HAVE_TDM 0
#endif

// ---------------------------------------------------------------------------
// Fragment loaders (wave32), contiguous k order shared by A and B (not the ISA table order).
// A (16x32 f16): lane = m + 16*h, m=row. halves i -> K = k0 + h*16 + i
// ---------------------------------------------------------------------------
static __device__ __forceinline__ v16h load_a_frag(const f16* __restrict__ base,
                                                   int lda, int k0) {
  int lane = threadIdx.x & 31;
  int m = lane & 15;
  int hl = lane >> 4;
  const f16* p = base + (size_t)m * lda + k0 + hl * 16;
  union { v16h v; v8h h[2]; } u;
  u.h[0] = *(const v8h*)(p);
  u.h[1] = *(const v8h*)(p + 8);
  return u.v;
}

// B (32x16 f16) with B[k][n] = colmat[n][k] (row n of colmat contiguous in k):
// lane = n + 16*h; halves i -> K = k0 + h*16 + i  => one contiguous 32B load.
static __device__ __forceinline__ v16h load_b_frag(const f16* __restrict__ colmat,
                                                   int ldb, int n0, int k0) {
  int lane = threadIdx.x & 31;
  int n = lane & 15;
  int hl = lane >> 4;
  const f16* p = colmat + (size_t)(n0 + n) * ldb + k0 + hl * 16;
  return *(const v16h*)p;
}

// Same, but from an LDS tile with row stride in halves (16B-aligned loads).
static __device__ __forceinline__ v16h load_b_frag_lds(const f16* tile, int stride,
                                                       int n0, int k0) {
  int lane = threadIdx.x & 31;
  int n = lane & 15;
  int hl = lane >> 4;
  const f16* p = tile + (n0 + n) * stride + k0 + hl * 16;
  union { v16h v; v8h h[2]; } u;
  u.h[0] = *(const v8h*)(p);
  u.h[1] = *(const v8h*)(p + 8);
  return u.v;
}

static __device__ __forceinline__ v8f wmma16(v16h a, v16h b, v8f c) {
  return __builtin_amdgcn_wmma_f32_16x16x32_f16(false, a, false, b,
                                                (short)0, c, false, false);
}

// ---------------------------------------------------------------------------
// 16-lane row max via DPP row_ror butterfly (rotations 1,2,4,8 cover all
// offsets within a row of 16).  Pure VALU: no ds_bpermute / dscnt stalls.
// ---------------------------------------------------------------------------
template <int CTRL>
static __device__ __forceinline__ float dpp_mov(float x) {
  int xi = __builtin_bit_cast(int, x);
  int r = __builtin_amdgcn_update_dpp(xi, xi, CTRL, 0xf, 0xf, true);
  return __builtin_bit_cast(float, r);
}
static __device__ __forceinline__ float rowmax16(float v) {
  v = fmaxf(v, dpp_mov<0x121>(v));  // row_ror:1
  v = fmaxf(v, dpp_mov<0x122>(v));  // row_ror:2
  v = fmaxf(v, dpp_mov<0x124>(v));  // row_ror:4
  v = fmaxf(v, dpp_mov<0x128>(v));  // row_ror:8
  return v;
}

// ---------------------------------------------------------------------------
// TDM: load a 32-row x 128-byte tile (row pitch 6144 B in global) into LDS,
// padded +16 B per 128 B so LDS rows land on a 144-B (36-dword) pitch.
// D# per CDNA5 ISA: group0 = {ctrl, lds_addr, gaddr},
// group1 = {flags, tensor dims, tile dims, strides}.
// ---------------------------------------------------------------------------
#if HAVE_TDM
static __device__ __forceinline__ void tdm_load_k(const f16* gsrc, unsigned lds_off) {
  unsigned long long ga = (unsigned long long)(uintptr_t)gsrc;
  u32x4 g0 = { 1u,                                   // count=1, user mode
               lds_off,                              // LDS byte address
               (unsigned)ga,                         // global_addr[31:0]
               (unsigned)((ga >> 32) & 0x01FFFFFFu)  // global_addr[56:32]
                 | (2u << 30) };                     // type=2 ("image")
  i32x8 g1 = { (int)((2u << 16)      // data_size = 4 B
                 | (1u << 20)        // pad_enable
                 | (4u << 22)        // pad_interval: 32 dwords
                 | (3u << 25)),      // pad_amount:   4 dwords (=16 B)
               0,                    // abar=0; tensor_dim0[15:0]=0 (dim0=1<<20)
               16,                   // tensor_dim0[31:16]=16; tensor_dim1 lo=0
               16 | (32 << 16),      // tensor_dim1 hi=16; tile_dim0 = 32 dwords
               32,                   // tile_dim1 = 32 rows; tile_dim2 = 0
               1536,                 // tensor_dim0_stride = 6144 B / 4
               0, 0 };
  i32x4 z4 = { 0, 0, 0, 0 };
#if __has_include(<hip/amd_detail/amd_gfx1250_TDM.h>)
  i32x8 z8 = { 0, 0, 0, 0, 0, 0, 0, 0 };
  __builtin_amdgcn_tensor_load_to_lds(g0, g1, z4, z4, z8, 0);   // 6-arg toolchain
#else
  __builtin_amdgcn_tensor_load_to_lds(g0, g1, z4, z4, 0);       // 5-arg toolchain
#endif
}
#endif

// ---------------------------------------------------------------------------
// fp32 -> f16 conversion, 4 elements/thread
// ---------------------------------------------------------------------------
__global__ void __launch_bounds__(256)
cvt_f32_f16(const float* __restrict__ s, f16* __restrict__ d, int n) {
  int i = (blockIdx.x * 256 + threadIdx.x) * 4;
  if (i < n) {
    float4 f = *(const float4*)(s + i);
    v4h h = { (f16)f.x, (f16)f.y, (f16)f.z, (f16)f.w };
    *(v4h*)(d + i) = h;
  }
}

// ---------------------------------------------------------------------------
// WMMA GEMM: OUT[m,n] = sum_k A[m,k] * W[n,k] (+ bias).  256 thr = 8 waves,
// wave tile 32x64 (2x4 WMMA tiles), block tile 64x256.
// ---------------------------------------------------------------------------
template <bool OUT_F16>
__global__ void __launch_bounds__(256)
wmma_gemm_xwt(const f16* __restrict__ A, const f16* __restrict__ W,
              void* __restrict__ OUT, const float* __restrict__ bias,
              int N, int K) {
  const int lane = threadIdx.x & 31;
  const int wave = threadIdx.x >> 5;
  const int wm = wave >> 2;            // 0..1
  const int wn = wave & 3;             // 0..3
  const int m0 = blockIdx.y * 64 + wm * 32;
  const int n0 = blockIdx.x * 256 + wn * 64;

  v8f acc[2][4] = {};

  for (int k0 = 0; k0 < K; k0 += 32) {
    if (k0 + 32 < K) {   // global_prefetch_b8 for the next K-slab
      __builtin_prefetch(A + (size_t)(m0 + (lane & 15)) * K + k0 + 32, 0, 0);
      __builtin_prefetch(W + (size_t)(n0 + (lane & 31)) * K + k0 + 32, 0, 0);
    }
    v16h a0 = load_a_frag(A + (size_t)m0 * K, K, k0);
    v16h a1 = load_a_frag(A + (size_t)(m0 + 16) * K, K, k0);
#pragma unroll
    for (int j = 0; j < 4; ++j) {
      v16h b = load_b_frag(W, K, n0 + j * 16, k0);
      acc[0][j] = wmma16(a0, b, acc[0][j]);
      acc[1][j] = wmma16(a1, b, acc[1][j]);
    }
  }

  const int hl = lane >> 4, nn = lane & 15;
#pragma unroll
  for (int i = 0; i < 2; ++i)
#pragma unroll
    for (int j = 0; j < 4; ++j)
#pragma unroll
      for (int r = 0; r < 8; ++r) {
        int row = m0 + i * 16 + r + 8 * hl;
        int col = n0 + j * 16 + nn;
        float v = acc[i][j][r];
        if constexpr (OUT_F16)
          ((f16*)OUT)[(size_t)row * N + col] = (f16)v;
        else
          ((float*)OUT)[(size_t)row * N + col] = v + bias[col];
      }
}

// ---------------------------------------------------------------------------
// Flash attention (causal).  Block = 4 waves = one 64-query tile of one
// (batch, head).  Each wave owns 16 query rows; key blocks of 32 streamed.
// qkv layout: [B*T, 3072] f16, sections Q | K | V, head h at offset h*64.
//
// K block staged once per block into LDS by the TDM (double-buffered,
// s_wait_tensorcnt + barrier), with TDM padding giving a conflict-free
// 144-B row pitch.  V staged transposed (vt[d][key], 80-B pitch).
// Softmax row-max via DPP; row-sum via ones-matrix WMMA.
// ---------------------------------------------------------------------------
#define VT_STRIDE 40   // halves (80 B)
#define KQ_STRIDE 72   // halves (144 B = 128 B row + 16 B TDM pad)

__global__ void __launch_bounds__(128)
flash_attn(const f16* __restrict__ qkv, f16* __restrict__ ctx) {
  const int lane = threadIdx.x & 31;
  const int wave = threadIdx.x >> 5;
  const int hl = lane >> 4, nn = lane & 15;
  const int q0  = blockIdx.x * 64;
  const int head = blockIdx.y;
  const int bz   = blockIdx.z;
  const int q0w = q0 + wave * 16;
  const size_t rowbase = (size_t)bz * SEQ;

  const f16* qp = qkv + rowbase * QKV3 + head * HDIM;
  const f16* kp = qp + EMBED;
  const f16* vp = qp + 2 * EMBED;

  __shared__ alignas(32) f16 vt_lds[64 * VT_STRIDE];     // V^T: 64 d x 32 keys
  __shared__ alignas(32) f16 p_lds[4][16 * 32];          // per-wave P relayout
  __shared__ alignas(32) f16 kq_lds[2][32 * KQ_STRIDE];  // K tiles (dbl-buffered)

  // Q fragments with 1/sqrt(D)=0.125 folded in (power of two -> exact in f16)
  v16h aq0 = load_a_frag(qp + (size_t)q0w * QKV3, QKV3, 0);
  v16h aq1 = load_a_frag(qp + (size_t)q0w * QKV3, QKV3, 32);
#pragma unroll
  for (int i = 0; i < 16; ++i) {
    aq0[i] = aq0[i] * (f16)0.125f;
    aq1[i] = aq1[i] * (f16)0.125f;
  }

  v16h vones;
#pragma unroll
  for (int i = 0; i < 16; ++i) vones[i] = (f16)1.0f;

  v8f o[4] = {};
  float mrow[8], lrow[8];
#pragma unroll
  for (int r = 0; r < 8; ++r) { mrow[r] = -3.0e38f; lrow[r] = 0.0f; }

  const int nblk = (q0 + 64) / 32;
  int buf = 0;

#if HAVE_TDM
  if (wave == 0)   // prime the pipeline: K block 0 -> buffer 0
    tdm_load_k(kp, (unsigned)(uintptr_t)&kq_lds[0][0]);
#endif

  for (int kb = 0; kb < nblk; ++kb) {
    __syncthreads();   // vt_lds / kq buffers free of previous readers
    {  // V^T stage: thread owns key-pair x 8 d; packed 32-bit LDS stores
      int t = threadIdx.x;
      int kp2 = (t & 15) * 2;        // key pair base (0,2,..,30)
      int d0 = (t >> 4) * 8;         // 8 d-values per thread
      const f16* s0p = vp + (size_t)(kb * 32 + kp2) * QKV3 + d0;
      v8h va = *(const v8h*)(s0p);
      v8h vb = *(const v8h*)(s0p + QKV3);
#pragma unroll
      for (int i = 0; i < 8; ++i) {
        union { f16 h[2]; unsigned u; } pk;
        pk.h[0] = va[i]; pk.h[1] = vb[i];
        *(unsigned*)(&vt_lds[(d0 + i) * VT_STRIDE + kp2]) = pk.u;
      }
    }
#if HAVE_TDM
    if (wave == 0) {
      __builtin_amdgcn_s_wait_tensorcnt(0);   // current K buffer landed
      if (kb + 1 < nblk)                       // prefetch next K block
        tdm_load_k(kp + (size_t)(kb + 1) * 32 * QKV3,
                   (unsigned)(uintptr_t)&kq_lds[buf ^ 1][0]);
    }
#else
    {  // fallback K stage: plain cooperative copy
      int t = threadIdx.x;
      int key = t >> 2, d0 = (t & 3) * 16;
      const f16* src = kp + (size_t)(kb * 32 + key) * QKV3 + d0;
      union { v16h v; v8h h[2]; } u;
      u.v = *(const v16h*)src;
      *(v8h*)(&kq_lds[0][key * KQ_STRIDE + d0])     = u.h[0];
      *(v8h*)(&kq_lds[0][key * KQ_STRIDE + d0 + 8]) = u.h[1];
    }
#endif
    __syncthreads();

    // S = Q * K^T  (16q x 32k per wave); K B-frags from the LDS tile
    const f16* kq = &kq_lds[buf][0];
    v8f s[2] = {};
#pragma unroll
    for (int j = 0; j < 2; ++j) {
      v16h b0 = load_b_frag_lds(kq, KQ_STRIDE, j * 16, 0);
      v16h b1 = load_b_frag_lds(kq, KQ_STRIDE, j * 16, 32);
      s[j] = wmma16(aq0, b0, s[j]);
      s[j] = wmma16(aq1, b1, s[j]);
    }
#if HAVE_TDM
    buf ^= 1;
#endif

    float sv0[8], sv1[8];
#pragma unroll
    for (int r = 0; r < 8; ++r) { sv0[r] = s[0][r]; sv1[r] = s[1][r]; }

    if (kb * 32 + 31 > q0w) {   // wave-uniform: boundary block, apply mask
#pragma unroll
      for (int r = 0; r < 8; ++r) {
        int row = q0w + r + 8 * hl;
        int c0 = kb * 32 + nn;
        if (c0 > row)      sv0[r] = -3.0e38f;
        if (c0 + 16 > row) sv1[r] = -3.0e38f;
      }
    }

    // online softmax: row-max via DPP butterfly, exp, rescale O
    float alpha[8], e0[8], e1[8];
#pragma unroll
    for (int r = 0; r < 8; ++r) {
      float mx = rowmax16(fmaxf(sv0[r], sv1[r]));
      float mnew = fmaxf(mrow[r], mx);
      alpha[r] = __expf(mrow[r] - mnew);
      e0[r] = __expf(sv0[r] - mnew);
      e1[r] = __expf(sv1[r] - mnew);
      mrow[r] = mnew;
#pragma unroll
      for (int j = 0; j < 4; ++j) o[j][r] *= alpha[r];
    }

    // stage P (f16) row-major 16x32 in per-wave LDS; same-wave DS ops are
    // in order on CDNA5, so the A-fragment reload below sees the stores.
    f16* pw = &p_lds[wave][0];
#pragma unroll
    for (int r = 0; r < 8; ++r) {
      int row = r + 8 * hl;
      pw[row * 32 + nn]      = (f16)e0[r];
      pw[row * 32 + 16 + nn] = (f16)e1[r];
    }
    v16h ap;
    {
      const f16* pp = pw + nn * 32 + hl * 16;
      union { v16h v; v8h h[2]; } u;
      u.h[0] = *(const v8h*)(pp);
      u.h[1] = *(const v8h*)(pp + 8);
      ap = u.v;
    }

    // row sums of P via ones-matrix WMMA: rs[r] = sum_k P[row,k] in all lanes
    v8f zc = {};
    v8f rs = wmma16(ap, vones, zc);

    // O += P * V: B-frag lane nn -> d = j*16+nn, halves = keys hl*16..+15,
    // contiguous in vt_lds -> two ds_load_b128 per fragment.
#pragma unroll
    for (int j = 0; j < 4; ++j) {
      const f16* vb = &vt_lds[(j * 16 + nn) * VT_STRIDE + hl * 16];
      union { v16h v; v8h h[2]; } u;
      u.h[0] = *(const v8h*)(vb);
      u.h[1] = *(const v8h*)(vb + 8);
      o[j] = wmma16(ap, u.v, o[j]);
    }

#pragma unroll
    for (int r = 0; r < 8; ++r) lrow[r] = lrow[r] * alpha[r] + rs[r];
  }

  // normalize and store f16 context [B*T, 1024]
#pragma unroll
  for (int r = 0; r < 8; ++r) {
    float inv = 1.0f / lrow[r];
    int row = q0w + r + 8 * hl;
#pragma unroll
    for (int j = 0; j < 4; ++j)
      ctx[(rowbase + row) * EMBED + head * HDIM + j * 16 + nn] =
          (f16)(o[j][r] * inv);
  }
}

// ---------------------------------------------------------------------------
// Host side
// ---------------------------------------------------------------------------
extern "C" void kernel_launch(void* const* d_in, const int* in_sizes, int n_in,
                              void* d_out, int out_size, void* d_ws, size_t ws_size,
                              hipStream_t stream) {
  const float* x      = (const float*)d_in[0];   // [4,2048,1024]
  const float* w_qkv  = (const float*)d_in[1];   // [3072,1024]
  const float* w_proj = (const float*)d_in[2];   // [1024,1024]
  const float* b_proj = (const float*)d_in[3];   // [1024]
  float* out = (float*)d_out;

  char* ws = (char*)d_ws;                        // ~92.3 MB used
  f16* xh   = (f16*)(ws);                        // 16.0 MiB
  f16* wh   = (f16*)(ws + 16777216);             //  6.0 MiB
  f16* wph  = (f16*)(ws + 23068672);             //  2.0 MiB
  f16* qkvh = (f16*)(ws + 25165824);             // 48.0 MiB
  f16* ctxh = (f16*)(ws + 75497472);             // 16.0 MiB

  const int nx  = MTOT * EMBED;        // 8388608
  const int nwq = QKV3 * EMBED;        // 3145728
  const int nwp = EMBED * EMBED;       // 1048576
  cvt_f32_f16<<<nx  / 1024, 256, 0, stream>>>(x,      xh,  nx);
  cvt_f32_f16<<<nwq / 1024, 256, 0, stream>>>(w_qkv,  wh,  nwq);
  cvt_f32_f16<<<nwp / 1024, 256, 0, stream>>>(w_proj, wph, nwp);

  // QKV projection: [8192,1024] x [3072,1024]^T -> f16 [8192,3072]
  dim3 g1(QKV3 / 256, MTOT / 64);
  wmma_gemm_xwt<true><<<g1, 256, 0, stream>>>(xh, wh, qkvh, nullptr, QKV3, EMBED);

  // causal flash attention -> f16 context [8192,1024]
  dim3 g2(SEQ / 64, NHEAD, BATCH);
  flash_attn<<<g2, 128, 0, stream>>>(qkvh, ctxh);

  // output projection + bias: [8192,1024] x [1024,1024]^T -> f32 d_out
  dim3 g3(EMBED / 256, MTOT / 64);
  wmma_gemm_xwt<false><<<g3, 256, 0, stream>>>(ctxh, wph, out, b_proj, EMBED, EMBED);
}
